// _AttentionSelf_12773232738460
// MI455X (gfx1250) — matrix operand order repaired and checked by reading, not compiled, not run on hardware
//
#include <hip/hip_runtime.h>

typedef __attribute__((ext_vector_type(16))) __bf16 v16bf;
typedef __attribute__((ext_vector_type(8)))  float  v8f;

#define BATCH 2
#define S_LEN 2048
#define D_DIM 2048
#define NHEAD 16
#define DKH   128

__device__ __forceinline__ unsigned short f2bf(float f) {
    unsigned int u = __float_as_uint(f);
    u += 0x7fffu + ((u >> 16) & 1u);     // round-to-nearest-even
    return (unsigned short)(u >> 16);
}

union FragBF {
    v16bf v;
    unsigned int d[8];
    uint4 q[2];
    unsigned short h[16];
};

__device__ __forceinline__ v8f wmma_bf16(v16bf a, v16bf b, v8f c) {
    return __builtin_amdgcn_wmma_f32_16x16x32_bf16(false, a, false, b, (short)0, c, false, false);
}

// Async global->LDS copy of 16 bytes per lane (GLOBAL_LOAD_ASYNC_TO_LDS_B128,
// tracked by ASYNCcnt; 'done' returns in order per wave).
__device__ __forceinline__ void async_copy_b128(const unsigned short* g, unsigned short* l) {
    unsigned lofs = (unsigned)(uintptr_t)(__attribute__((address_space(3))) unsigned short*)l;
    asm volatile("global_load_async_to_lds_b128 %0, %1, off" :: "v"(lofs), "v"(g) : "memory");
}
#define WAIT_ASYNC(n) asm volatile("s_wait_asynccnt " #n ::: "memory")

// ---------------------------------------------------------------------------
// fp32 -> bf16 bulk conversion (8 elements / thread)
// ---------------------------------------------------------------------------
__global__ __launch_bounds__(256)
void cvt_bf16_kernel(const float* __restrict__ in, unsigned short* __restrict__ out, int n8)
{
    int idx = blockIdx.x * 256 + threadIdx.x;
    if (idx < n8) {
        const float4* p = (const float4*)in + (size_t)idx * 2;
        float4 a = p[0], b = p[1];
        uint4 o;
        o.x = (unsigned)f2bf(a.x) | ((unsigned)f2bf(a.y) << 16);
        o.y = (unsigned)f2bf(a.z) | ((unsigned)f2bf(a.w) << 16);
        o.z = (unsigned)f2bf(b.x) | ((unsigned)f2bf(b.y) << 16);
        o.w = (unsigned)f2bf(b.z) | ((unsigned)f2bf(b.w) << 16);
        *(uint4*)(out + (size_t)idx * 8) = o;
    }
}

// ---------------------------------------------------------------------------
// GEMM: Y[m,n] = X[m,:] . W[n,:] + bias[n]   (X, W already bf16)
// M = BATCH*S_LEN = 4096, N = K = 2048.
// Double-buffered LDS, tiles staged with global_load_async_to_lds_b128;
// ASYNCcnt-based pipelining (8 async ops per wave per buffer).
// mode 0: out bf16 [B,H,S,DK]      (Q, K)
// mode 1: out bf16 [B,H,DK,S]      (V transposed, for contiguous attention reads)
// mode 2: out f32  [M,N]           (final output projection)
// ---------------------------------------------------------------------------
#define BM 128
#define BN 128
#define BK 64
#define NKT (D_DIM / BK)
#define LDSS 72   // padded stride (halves) to stagger LDS banks

__global__ __launch_bounds__(256)
void gemm_bf16_kernel(const unsigned short* __restrict__ X,
                      const unsigned short* __restrict__ Wb,
                      const float* __restrict__ bias,
                      void* __restrict__ Out, int mode)
{
    __shared__ __align__(16) unsigned short lds_a[2][BM * LDSS];
    __shared__ __align__(16) unsigned short lds_b[2][BN * LDSS];

    const int t    = threadIdx.x;
    const int wave = t >> 5;
    const int lane = t & 31;
    const int g    = lane >> 4;
    const int r    = lane & 15;
    const int wm   = wave >> 1;   // 0..3  (rows of 32)
    const int wn   = wave & 1;    // 0..1  (cols of 64)
    const int tileM = blockIdx.y;
    const int tileN = blockIdx.x;

    v8f acc[2][4];
    for (int mf = 0; mf < 2; ++mf)
        for (int nf = 0; nf < 4; ++nf)
            for (int v = 0; v < 8; ++v) acc[mf][nf][v] = 0.0f;

    // issue async copies for one k-tile into LDS buffer `buf` (8 ops / wave)
    auto issueCopies = [&](int buf, int kt) {
        const int k0 = kt * BK;
        for (int i = 0; i < 4; ++i) {                 // X tile: 128 x 64 halves
            int lin = t + i * 256, row = lin >> 3, c = lin & 7;
            async_copy_b128(X + (size_t)(tileM * BM + row) * D_DIM + k0 + c * 8,
                            lds_a[buf] + row * LDSS + c * 8);
        }
        for (int i = 0; i < 4; ++i) {                 // W tile: 128 x 64 halves
            int lin = t + i * 256, row = lin >> 3, c = lin & 7;
            async_copy_b128(Wb + (size_t)(tileN * BN + row) * D_DIM + k0 + c * 8,
                            lds_b[buf] + row * LDSS + c * 8);
        }
    };

    issueCopies(0, 0);

    for (int kt = 0; kt < NKT; ++kt) {
        const int cur = kt & 1;

        if (kt + 1 < NKT) {
            issueCopies(cur ^ 1, kt + 1);             // deepen the pipeline
            if (kt + 2 < NKT) {                       // global_prefetch_b8, 2 ahead
                int prow = t >> 1, phalf = (t & 1) * 32, pk = (kt + 2) * BK;
                __builtin_prefetch(X + (size_t)(tileM * BM + prow) * D_DIM + pk + phalf, 0, 1);
                __builtin_prefetch(Wb + (size_t)(tileN * BN + prow) * D_DIM + pk + phalf, 0, 1);
            }
            WAIT_ASYNC(8);                            // oldest buffer (cur) complete
        } else {
            WAIT_ASYNC(0);
        }
        __syncthreads();                              // cur tiles visible to all waves

        for (int ks = 0; ks < BK / 32; ++ks) {
            FragBF a[2], b[4];
            for (int mf = 0; mf < 2; ++mf) {
                int row = wm * 32 + mf * 16 + r;      // A layout: M = lane%16
                const unsigned int* pr = (const unsigned int*)(lds_a[cur] + row * LDSS);
                const int base = ks * 16 + 4 * g;
                for (int j = 0; j < 4; ++j) {
                    a[mf].d[j]     = pr[base + j];      // K = ks*32 + 2j+8g, +1
                    a[mf].d[4 + j] = pr[base + 8 + j];  // K = ks*32 + 16+2j+8g, +1
                }
            }
            for (int nf = 0; nf < 4; ++nf) {
                int col = wn * 64 + nf * 16 + r;      // B layout: N = lane%16
                const uint4* pb = (const uint4*)(lds_b[cur] + col * LDSS + ks * 32 + 8 * g);
                b[nf].q[0] = pb[0];                   // K = ks*32 + 8g + 0..7
                b[nf].q[1] = pb[2];                   // K = ks*32 + 16 + 8g + 0..7
            }
            for (int mf = 0; mf < 2; ++mf)
                for (int nf = 0; nf < 4; ++nf)
                    acc[mf][nf] = wmma_bf16(a[mf].v, b[nf].v, acc[mf][nf]);
        }

        __syncthreads();   // all waves done reading cur before next iter overwrites it
    }

    // epilogue: C/D layout row M = v + 8g, col N = r
    for (int mf = 0; mf < 2; ++mf) {
        for (int nf = 0; nf < 4; ++nf) {
            const int n = tileN * BN + wn * 64 + nf * 16 + r;
            const float bn = bias[n];
            for (int v = 0; v < 8; ++v) {
                const int m = tileM * BM + wm * 32 + mf * 16 + v + 8 * g;
                const float val = acc[mf][nf][v] + bn;
                if (mode == 0) {
                    const int b_ = m >> 11, s_ = m & 2047, h_ = n >> 7, dk_ = n & 127;
                    ((unsigned short*)Out)[((((size_t)b_ * NHEAD + h_) * S_LEN + s_) << 7) + dk_] = f2bf(val);
                } else if (mode == 1) {
                    const int b_ = m >> 11, s_ = m & 2047, h_ = n >> 7, dk_ = n & 127;
                    ((unsigned short*)Out)[((((size_t)b_ * NHEAD + h_) * DKH + dk_) << 11) + s_] = f2bf(val);
                } else {
                    ((float*)Out)[(size_t)m * D_DIM + n] = val;
                }
            }
        }
    }
}

// ---------------------------------------------------------------------------
// Gate: G[b,s,h] = sigmoid(states[b,s,:] . Wg[h,:] + bg[h])
// ---------------------------------------------------------------------------
__global__ __launch_bounds__(256)
void gate_kernel(const float* __restrict__ states, const float* __restrict__ Wg,
                 const float* __restrict__ bg, float* __restrict__ G)
{
    const int row = blockIdx.x;
    const int t = threadIdx.x;
    const int h = t >> 4, j = t & 15;   // same-h threads share a 16-lane half
    const float* x  = states + (size_t)row * D_DIM;
    const float* wg = Wg + (size_t)h * D_DIM;
    float s = 0.f;
    for (int k = 0; k < 128; ++k) s += x[j * 128 + k] * wg[j * 128 + k];
    for (int mask = 1; mask <= 8; mask <<= 1) s += __shfl_xor(s, mask, 32);
    if (j == 0) G[(size_t)row * NHEAD + h] = 1.0f / (1.0f + __expf(-(s + bg[h])));
}

// ---------------------------------------------------------------------------
// Flash attention with analytic causal ALiBi bias.
// Block = 4 waves, 64 q-rows; each wave owns 16 q-rows.
// K/V tiles staged once per block into LDS via global_load_async_to_lds_b128.
// Q,K bf16 [B,H,S,DK]; V bf16 transposed [B,H,DK,S]; output gated bf16 [B,S,D].
// ---------------------------------------------------------------------------
#define KSTR 136   // ldsK row stride in halves (128 + 8 pad)
#define VSTR 72    // ldsV row stride in halves (64 + 8 pad)

__global__ __launch_bounds__(128)
void attn_kernel(const unsigned short* __restrict__ Q,
                 const unsigned short* __restrict__ K,
                 const unsigned short* __restrict__ Vt,
                 const float* __restrict__ G,
                 const float* __restrict__ head_scale,
                 unsigned short* __restrict__ Og)
{
    __shared__ __align__(16) unsigned short ldsK[64 * KSTR];     // [key 64][dk 128]
    __shared__ __align__(16) unsigned short ldsV[128 * VSTR];    // [dk 128][key 64]
    __shared__ __align__(16) unsigned short Pl[4][16][72];       // per-wave P tile

    const int t = threadIdx.x;
    const int w = t >> 5, lane = t & 31, g = lane >> 4, r = lane & 15;
    const int qt = blockIdx.x;
    const int bh = blockIdx.y;
    const int b  = bh >> 4, h = bh & 15;
    const int q0 = qt * 64 + w * 16;
    const float hs = head_scale[h];
    const float key_scale = 0.08838834764831845f;   // 128^-0.5

    const unsigned short* Qb = Q  + (size_t)bh * S_LEN * DKH;
    const unsigned short* Kb = K  + (size_t)bh * S_LEN * DKH;
    const unsigned short* Vb = Vt + (size_t)bh * DKH * S_LEN;

    // Q A-fragments (dk in chunks of 32), loaded once.
    FragBF qa[4];
    {
        const int qrow = q0 + r;                     // A layout: M = lane%16
        for (int dkc = 0; dkc < 4; ++dkc) {
            qa[dkc].q[0] = *(const uint4*)(Qb + (size_t)qrow * DKH + dkc * 32 + 8 * g);
            qa[dkc].q[1] = *(const uint4*)(Qb + (size_t)qrow * DKH + dkc * 32 + 16 + 8 * g);
        }
    }

    v8f acc[8];
    float mrun[8], lrun[8];
    for (int i = 0; i < 8; ++i) {
        mrun[i] = -1e30f; lrun[i] = 0.f;
        for (int v = 0; v < 8; ++v) acc[i][v] = 0.f;
    }

    for (int kt = 0; kt <= qt; ++kt) {              // causal: skip fully-masked tiles
        __syncthreads();                            // previous iteration's reads done
        // stage K tile: 64 rows x 128 halves, 8x b128 per thread
        for (int i = 0; i < 8; ++i) {
            int lin = t + i * 128, row = lin >> 4, c = lin & 15;
            async_copy_b128(Kb + (size_t)(kt * 64 + row) * DKH + c * 8,
                            ldsK + row * KSTR + c * 8);
        }
        // stage V tile: 128 dk-rows x 64 halves
        for (int i = 0; i < 8; ++i) {
            int lin = t + i * 128, row = lin >> 3, c = lin & 7;
            async_copy_b128(Vb + (size_t)row * S_LEN + kt * 64 + c * 8,
                            ldsV + row * VSTR + c * 8);
        }
        WAIT_ASYNC(0);
        __syncthreads();                            // tiles visible to all waves

        // S = Q @ K^T  (16 x 64 per wave)
        v8f sc[4];
        for (int cf = 0; cf < 4; ++cf)
            for (int v = 0; v < 8; ++v) sc[cf][v] = 0.f;
        for (int cf = 0; cf < 4; ++cf) {
            const unsigned short* kp = ldsK + (cf * 16 + r) * KSTR;  // B layout: N = lane%16
            for (int dkc = 0; dkc < 4; ++dkc) {
                FragBF kb;
                const uint4* pk4 = (const uint4*)(kp + dkc * 32 + 8 * g);
                kb.q[0] = pk4[0];
                kb.q[1] = pk4[2];
                sc[cf] = wmma_bf16(qa[dkc].v, kb.v, sc[cf]);
            }
        }
        // scale + analytic ALiBi bias, row max
        float rmax[8];
        for (int v = 0; v < 8; ++v) rmax[v] = -3.0e38f;
        for (int cf = 0; cf < 4; ++cf) {
            const int kcol = kt * 64 + cf * 16 + r;
            for (int v = 0; v < 8; ++v) {
                const int qrow = q0 + v + 8 * g;
                float s = sc[cf][v] * key_scale
                        + ((kcol <= qrow) ? (float)(kcol - qrow) * hs : -1e9f * hs);
                sc[cf][v] = s;
                rmax[v] = fmaxf(rmax[v], s);
            }
        }
        for (int v = 0; v < 8; ++v)
            for (int mask = 1; mask <= 8; mask <<= 1)    // reduce within 16-lane half
                rmax[v] = fmaxf(rmax[v], __shfl_xor(rmax[v], mask, 32));

        // online softmax rescale
        float rsum[8];
        for (int v = 0; v < 8; ++v) {
            const float mnew  = fmaxf(mrun[v], rmax[v]);
            const float alpha = __expf(mrun[v] - mnew);
            mrun[v]  = mnew;
            lrun[v] *= alpha;
            for (int nf = 0; nf < 8; ++nf) acc[nf][v] *= alpha;
            rsum[v] = 0.f;
        }
        // P = exp(S - m), stash into LDS (C-layout -> A-layout restage)
        for (int cf = 0; cf < 4; ++cf) {
            for (int v = 0; v < 8; ++v) {
                const float p = __expf(sc[cf][v] - mrun[v]);
                rsum[v] += p;
                Pl[w][v + 8 * g][cf * 16 + r] = f2bf(p);
            }
        }
        for (int v = 0; v < 8; ++v) {
            for (int mask = 1; mask <= 8; mask <<= 1)
                rsum[v] += __shfl_xor(rsum[v], mask, 32);
            lrun[v] += rsum[v];
        }

        // O += P @ V  (V tile in LDS: contiguous key-dim per dk row)
        for (int kf = 0; kf < 2; ++kf) {
            FragBF pf;
            const unsigned int* pr = (const unsigned int*)&Pl[w][r][0];
            const int base = kf * 16 + 4 * g;
            for (int j = 0; j < 4; ++j) {
                pf.d[j]     = pr[base + j];
                pf.d[4 + j] = pr[base + 8 + j];
            }
            for (int nf = 0; nf < 8; ++nf) {
                FragBF vb;
                const uint4* pv4 = (const uint4*)(ldsV + (nf * 16 + r) * VSTR + kf * 32 + 8 * g);
                vb.q[0] = pv4[0];
                vb.q[1] = pv4[2];
                acc[nf] = wmma_bf16(pf.v, vb.v, acc[nf]);
            }
        }
    }

    // epilogue: normalize, gate, store bf16 [B,S,D]
    float inv[8], gate[8];
    for (int v = 0; v < 8; ++v) {
        inv[v] = 1.0f / lrun[v];
        const int qrow = q0 + v + 8 * g;
        gate[v] = G[((size_t)b * S_LEN + qrow) * NHEAD + h];
    }
    for (int nf = 0; nf < 8; ++nf) {
        for (int v = 0; v < 8; ++v) {
            const int qrow = q0 + v + 8 * g;
            const float o = acc[nf][v] * inv[v] * gate[v];
            Og[((size_t)b * S_LEN + qrow) * D_DIM + h * DKH + nf * 16 + r] = f2bf(o);
        }
    }
}

// ---------------------------------------------------------------------------
extern "C" void kernel_launch(void* const* d_in, const int* in_sizes, int n_in,
                              void* d_out, int out_size, void* d_ws, size_t ws_size,
                              hipStream_t stream)
{
    const float* states     = (const float*)d_in[0];
    // d_in[1] = bias_mask (unused: computed analytically)
    const float* head_scale = (const float*)d_in[2];
    const float* Wq = (const float*)d_in[3];  const float* bq = (const float*)d_in[4];
    const float* Wk = (const float*)d_in[5];  const float* bk = (const float*)d_in[6];
    const float* Wv = (const float*)d_in[7];  const float* bv = (const float*)d_in[8];
    const float* Wg = (const float*)d_in[9];  const float* bg = (const float*)d_in[10];
    const float* Wo = (const float*)d_in[11]; const float* bo = (const float*)d_in[12];

    const size_t stElems  = (size_t)BATCH * S_LEN * D_DIM;         // 8M (states / Og)
    const size_t wElems   = (size_t)D_DIM * D_DIM;                 // 4M (one weight)
    const size_t qkvElems = (size_t)BATCH * NHEAD * S_LEN * DKH;   // 8M

    unsigned short* SB  = (unsigned short*)d_ws;   // states bf16; reused as Og after attn
    unsigned short* WSb = SB + stElems;            // rotating weight slot (bf16)
    unsigned short* Qb  = WSb + wElems;
    unsigned short* Kb  = Qb + qkvElems;
    unsigned short* Vt  = Kb + qkvElems;
    float*          G   = (float*)(Vt + qkvElems);

    const int stBlocks = (int)(stElems / 8 / 256);                 // 4096
    const int wBlocks  = (int)(wElems / 8 / 256);                  // 2048
    dim3 gGrid(D_DIM / BN, (BATCH * S_LEN) / BM);                  // 16 x 32

    cvt_bf16_kernel<<<stBlocks, 256, 0, stream>>>(states, SB, (int)(stElems / 8));

    cvt_bf16_kernel<<<wBlocks, 256, 0, stream>>>(Wq, WSb, (int)(wElems / 8));
    gemm_bf16_kernel<<<gGrid, 256, 0, stream>>>(SB, WSb, bq, Qb, 0);

    cvt_bf16_kernel<<<wBlocks, 256, 0, stream>>>(Wk, WSb, (int)(wElems / 8));
    gemm_bf16_kernel<<<gGrid, 256, 0, stream>>>(SB, WSb, bk, Kb, 0);

    cvt_bf16_kernel<<<wBlocks, 256, 0, stream>>>(Wv, WSb, (int)(wElems / 8));
    gemm_bf16_kernel<<<gGrid, 256, 0, stream>>>(SB, WSb, bv, Vt, 1);

    gate_kernel<<<BATCH * S_LEN, 256, 0, stream>>>(states, Wg, bg, G);

    unsigned short* Og = SB;   // states_bf16 dead after V projection; reuse as Og
    attn_kernel<<<dim3(S_LEN / 64, BATCH * NHEAD), 128, 0, stream>>>(Qb, Kb, Vt, G, head_scale, Og);

    cvt_bf16_kernel<<<wBlocks, 256, 0, stream>>>(Wo, WSb, (int)(wElems / 8));
    gemm_bf16_kernel<<<gGrid, 256, 0, stream>>>(Og, WSb, bo, d_out, 2);
}
